// QuadraticSplineLayer_34574486733366
// MI455X (gfx1250) — compile-verified, matrix operand order checked by reading, not run on hardware
//
#include <hip/hip_runtime.h>
#include <hip/hip_bf16.h>

// ---------------- problem constants ----------------
#define BDIM   2048
#define SIZE   2048
#define KBINS  8
#define HID    1024
#define OUTN   (SIZE * (2*KBINS + 1))   // 34816
#define EPSF   1e-6f
#define INV_PI 0.31830988618379067154f

typedef __attribute__((ext_vector_type(16))) __bf16 v16bf;
typedef __attribute__((ext_vector_type(8)))  __bf16 v8bf;
typedef __attribute__((ext_vector_type(8)))  float  v8f;

union FragBF { v16bf v; v8bf h[2]; };

__device__ __forceinline__ float ftanh(float x) {
    float e = __expf(2.0f * x);
    return 1.0f - 2.0f / (e + 1.0f);
}

// ---- CDNA5 async global->LDS copy (16B per lane), tracked by ASYNCcnt ----
__device__ __forceinline__ void async_b128(void* lds, const void* g) {
    asm volatile("global_load_async_to_lds_b128 %0, %1, off"
                 :: "v"((unsigned int)(uintptr_t)lds), "v"(g)
                 : "memory");
}
__device__ __forceinline__ void wait_async0() {
    asm volatile("s_wait_asynccnt 0" ::: "memory");
}

// ---------------- init: copy log_density -> out tail, zero stats ----------------
__global__ void init_kernel(const float* __restrict__ ld_in,
                            float* __restrict__ out_ld,
                            float* __restrict__ stats) {
    int i = blockIdx.x * blockDim.x + threadIdx.x;
    if (i < BDIM) out_ld[i] = ld_in[i];
    if (i < 2)    stats[i] = 0.0f;
}

// ---------------- global sum / sumsq over x_passive ----------------
__global__ __launch_bounds__(256) void stats_kernel(const float* __restrict__ xp,
                                                    float* __restrict__ stats) {
    const int n4 = (BDIM * SIZE) / 4;
    float s = 0.0f, s2 = 0.0f;
    for (int i = blockIdx.x * blockDim.x + threadIdx.x; i < n4;
         i += gridDim.x * blockDim.x) {
        float4 v = ((const float4*)xp)[i];
        s  += v.x + v.y + v.z + v.w;
        s2 += v.x*v.x + v.y*v.y + v.z*v.z + v.w*v.w;
    }
    __shared__ float rs[256], rs2[256];
    int t = threadIdx.x;
    rs[t] = s; rs2[t] = s2;
    __syncthreads();
    for (int off = 128; off > 0; off >>= 1) {
        if (t < off) { rs[t] += rs[t + off]; rs2[t] += rs2[t + off]; }
        __syncthreads();
    }
    if (t == 0) {
        atomicAdd(&stats[0], rs[0]);
        atomicAdd(&stats[1], rs2[0]);
    }
}

// ---------------- standardize (ddof=1) + cvt to bf16 ----------------
__global__ __launch_bounds__(256) void standardize_kernel(const float* __restrict__ xp,
                                                          const float* __restrict__ stats,
                                                          __bf16* __restrict__ xsb) {
    const float N  = (float)(BDIM * SIZE);
    float s  = stats[0];
    float s2 = stats[1];
    float mean = s / N;
    float var  = (s2 - s * mean) / (N - 1.0f);
    float rstd = rsqrtf(var);
    const int n4 = (BDIM * SIZE) / 4;
    for (int i = blockIdx.x * blockDim.x + threadIdx.x; i < n4;
         i += gridDim.x * blockDim.x) {
        float4 v = ((const float4*)xp)[i];
        __bf16* d = xsb + (size_t)i * 4;
        d[0] = (__bf16)((v.x - mean) * rstd);
        d[1] = (__bf16)((v.y - mean) * rstd);
        d[2] = (__bf16)((v.z - mean) * rstd);
        d[3] = (__bf16)((v.w - mean) * rstd);
    }
}

// ---------------- tiled transpose + f32 -> bf16: dst[C][R] = (bf16)src[R][C] ----------------
__global__ __launch_bounds__(256) void transpose_cvt_kernel(const float* __restrict__ src,
                                                            __bf16* __restrict__ dst,
                                                            int R, int C) {
    __shared__ float tile[32][33];
    const int tx = threadIdx.x & 31, ty = threadIdx.x >> 5;   // 32 x 8
    const int c0 = blockIdx.x * 32, r0 = blockIdx.y * 32;
    #pragma unroll
    for (int i = ty; i < 32; i += 8)
        tile[i][tx] = src[(size_t)(r0 + i) * C + c0 + tx];
    __syncthreads();
    #pragma unroll
    for (int i = ty; i < 32; i += 8)
        dst[(size_t)(c0 + i) * R + r0 + tx] = (__bf16)tile[tx][i];
}

// ---------------- B fragment load helper (col-major LDS [n][k]) ----------------
__device__ __forceinline__ FragBF load_fragB(const __bf16* Bs, int t, int rowl, int half) {
    FragBF f;
    f.h[0] = *(const v8bf*)&Bs[(t * 16 + rowl) * 32 + half * 8];
    f.h[1] = *(const v8bf*)&Bs[(t * 16 + rowl) * 32 + 16 + half * 8];
    return f;
}

// ---------------- GEMM1: h = tanh(xs @ w1 + b1), bf16 out ----------------
// Tile 128(M) x 128(N), K = SIZE. Async double-buffered LDS staging.
__global__ __launch_bounds__(256) void gemm1_kernel(const __bf16* __restrict__ xsb,
                                                    const __bf16* __restrict__ w1t, // [HID][SIZE]
                                                    const float* __restrict__ b1,
                                                    __bf16* __restrict__ hb) {
    __shared__ __align__(32) char smem[32768];

    const int tid  = threadIdx.x;
    const int w    = tid >> 5, lane = tid & 31;
    const int rowl = lane & 15, half = lane >> 4;
    const int tileM = blockIdx.y * 128;
    const int tileN = blockIdx.x * 128;

    v8f acc[8] = {};

    auto stage = [&](int buf, int kk) {
        __bf16* Ab = (__bf16*)&smem[buf * 8192];
        __bf16* Bb = (__bf16*)&smem[16384 + buf * 8192];
        #pragma unroll
        for (int i = 0; i < 2; ++i) {           // A: 128 rows x 32 k, 512 x 16B chunks
            int c = tid + i * 256;
            int r = c >> 2, sub = c & 3;
            async_b128(&Ab[r * 32 + sub * 8],
                       xsb + (size_t)(tileM + r) * SIZE + kk + sub * 8);
        }
        #pragma unroll
        for (int i = 0; i < 2; ++i) {           // B: 128 n-rows x 32 k (pre-transposed)
            int c = tid + i * 256;
            int n = c >> 2, sub = c & 3;
            async_b128(&Bb[n * 32 + sub * 8],
                       w1t + (size_t)(tileN + n) * SIZE + kk + sub * 8);
        }
    };

    const int NK = SIZE / 32;
    stage(0, 0);
    #pragma unroll 1
    for (int it = 0; it < NK; ++it) {
        wait_async0();
        __syncthreads();
        if (it + 1 < NK) stage((it + 1) & 1, (it + 1) * 32);

        const int par = it & 1;
        const __bf16* As = (const __bf16*)&smem[par * 8192];
        const __bf16* Bs = (const __bf16*)&smem[16384 + par * 8192];
        FragBF fa;
        fa.h[0] = *(const v8bf*)&As[(w * 16 + rowl) * 32 + half * 8];
        fa.h[1] = *(const v8bf*)&As[(w * 16 + rowl) * 32 + 16 + half * 8];
        #pragma unroll
        for (int t = 0; t < 8; ++t) {
            FragBF fb = load_fragB(Bs, t, rowl, half);
            acc[t] = __builtin_amdgcn_wmma_f32_16x16x32_bf16(
                false, fa.v, false, fb.v, (short)0, acc[t], false, false);
        }
    }

    float b1v[8];
    #pragma unroll
    for (int t = 0; t < 8; ++t) b1v[t] = b1[tileN + t * 16 + rowl];
    #pragma unroll
    for (int t = 0; t < 8; ++t) {
        #pragma unroll
        for (int j = 0; j < 8; ++j) {
            int grow = tileM + w * 16 + j + half * 8;   // C layout: vgpr j -> M=j / j+8
            int gcol = tileN + t * 16 + rowl;
            hb[(size_t)grow * HID + gcol] = (__bf16)ftanh(acc[t][j] + b1v[t]);
        }
    }
}

// ---------------- spline evaluation on one 17-vector (in LDS) ----------------
__device__ __forceinline__ void spline_eval(const float* net, float x,
                                            float* phi, float* logterm) {
    float eh[9];
    #pragma unroll
    for (int j = 0; j < 9; ++j) eh[j] = __expf(net[j]);
    float wmax = net[9];
    #pragma unroll
    for (int i = 1; i < 8; ++i) wmax = fmaxf(wmax, net[9 + i]);
    float wn[8];
    float wsum = 0.0f;
    #pragma unroll
    for (int i = 0; i < 8; ++i) { wn[i] = __expf(net[9 + i] - wmax); wsum += wn[i]; }
    float invw = 1.0f / wsum;
    float denom = 0.0f;
    #pragma unroll
    for (int i = 0; i < 8; ++i) { wn[i] *= invw; denom += 0.5f * wn[i] * (eh[i] + eh[i + 1]); }
    float invd = 1.0f / denom;
    float hn[9];
    #pragma unroll
    for (int j = 0; j < 9; ++j) hn[j] = eh[j] * invd;

    float xk[9];
    xk[0] = -EPSF;
    #pragma unroll
    for (int i = 0; i < 8; ++i) xk[i + 1] = xk[i] + wn[i];
    int cnt = 0;
    #pragma unroll
    for (int i = 0; i < 9; ++i) cnt += (xk[i] < x) ? 1 : 0;
    int k = min(max(cnt - 1, 0), KBINS - 1);

    float w_k = wn[0], h_k = hn[0], h_k1 = hn[1], x_km1 = xk[0], phi_km1 = 0.0f;
    float pacc = 0.0f;
    #pragma unroll
    for (int i = 0; i < 8; ++i) {
        if (i == k) { w_k = wn[i]; h_k = hn[i]; h_k1 = hn[i + 1]; x_km1 = xk[i]; phi_km1 = pacc; }
        pacc += 0.5f * wn[i] * (hn[i] + hn[i + 1]);
    }
    float alpha = (x - x_km1) / w_k;
    *phi = phi_km1 + alpha * h_k * w_k + 0.5f * alpha * alpha * (h_k1 - h_k) * w_k;
    *logterm = __logf(h_k + alpha * (h_k1 - h_k));
}

// ---------------- GEMM2 fused with spline epilogue ----------------
// Tile 128(M) x 272(N = 16 spline groups), K = HID. Async double-buffered staging.
// Epilogue LDS strip aliases the staging buffers (dead after the k-loop).
__global__ __launch_bounds__(256) void gemm2_spline_kernel(const __bf16* __restrict__ hb,
                                                           const __bf16* __restrict__ w2t, // [OUTN][HID]
                                                           const float* __restrict__ b2,
                                                           const float* __restrict__ x_in,
                                                           float* __restrict__ out) {
    __shared__ __align__(32) char smem[51200];
    // layout during k-loop: A0 @0 (8K), A1 @8192 (8K), B0 @16384 (17K), B1 @33792 (17K)
    // layout after k-loop:  ep @0 (34816 B = 8 waves x 4x272 f32), rowacc @34816 (512 B)

    const int tid  = threadIdx.x;
    const int w    = tid >> 5, lane = tid & 31;
    const int rowl = lane & 15, half = lane >> 4;
    const int tileM  = blockIdx.y * 128;
    const int tileN  = blockIdx.x * 272;
    const int sgBase = blockIdx.x * 16;

    v8f acc[17] = {};

    auto stage = [&](int buf, int kk) {
        __bf16* Ab = (__bf16*)&smem[buf * 8192];
        __bf16* Bb = (__bf16*)&smem[16384 + buf * 17408];
        #pragma unroll
        for (int i = 0; i < 2; ++i) {           // A: 128 rows x 32 k
            int c = tid + i * 256;
            int r = c >> 2, sub = c & 3;
            async_b128(&Ab[r * 32 + sub * 8],
                       hb + (size_t)(tileM + r) * HID + kk + sub * 8);
        }
        #pragma unroll
        for (int i = 0; i < 5; ++i) {           // B: 272 n-rows x 32 k = 1088 chunks (pad to 1280)
            int c = tid + i * 256; if (c > 1087) c = 1087;
            int n = c >> 2, sub = c & 3;
            async_b128(&Bb[n * 32 + sub * 8],
                       w2t + (size_t)(tileN + n) * HID + kk + sub * 8);
        }
    };

    const int NK = HID / 32;
    stage(0, 0);
    #pragma unroll 1
    for (int it = 0; it < NK; ++it) {
        wait_async0();
        __syncthreads();
        if (it + 1 < NK) stage((it + 1) & 1, (it + 1) * 32);

        const int par = it & 1;
        const __bf16* As = (const __bf16*)&smem[par * 8192];
        const __bf16* Bs = (const __bf16*)&smem[16384 + par * 17408];
        FragBF fa;
        fa.h[0] = *(const v8bf*)&As[(w * 16 + rowl) * 32 + half * 8];
        fa.h[1] = *(const v8bf*)&As[(w * 16 + rowl) * 32 + 16 + half * 8];
        #pragma unroll
        for (int t = 0; t < 17; ++t) {
            FragBF fb = load_fragB(Bs, t, rowl, half);
            acc[t] = __builtin_amdgcn_wmma_f32_16x16x32_bf16(
                false, fa.v, false, fb.v, (short)0, acc[t], false, false);
        }
    }

    __syncthreads();                 // staging buffers die; ep/rowacc regions become live

    // -------- epilogue: bias + tanh + spline --------
    float* ep     = (float*)&smem[0];
    float* rowacc = (float*)&smem[34816];

    if (tid < 128) rowacc[tid] = 0.0f;

    float b2v[17];
    #pragma unroll
    for (int t = 0; t < 17; ++t) b2v[t] = b2[tileN + t * 16 + rowl];

    float* wreg = &ep[w * 4 * 272];

    #pragma unroll
    for (int chunk = 0; chunk < 4; ++chunk) {
        const int i0 = chunk * 2;    // accumulator vgprs i0, i0+1 -> rows {i0,i0+1,i0+8,i0+9}
        #pragma unroll
        for (int u = 0; u < 2; ++u) {
            int rr = u + 2 * half;   // region row for this lane's half
            #pragma unroll
            for (int t = 0; t < 17; ++t) {
                float v = ftanh(acc[t][i0 + u] + b2v[t]);
                wreg[rr * 272 + t * 16 + rowl] = v;
            }
        }
        __syncthreads();
        #pragma unroll
        for (int q = 0; q < 2; ++q) {   // 4 rows x 16 groups = 64 pairs, 2 per lane
            int p  = lane + 32 * q;
            int rr = p >> 4;            // 0..3
            int sg = p & 15;
            int row16 = (rr < 2) ? (i0 + rr) : (i0 + 8 + (rr - 2));
            int grow  = tileM + w * 16 + row16;
            int gs    = sgBase + sg;
            float x = x_in[(size_t)grow * SIZE + gs] * INV_PI;
            float phi, lt;
            spline_eval(&wreg[rr * 272 + sg * 17], x, &phi, &lt);
            out[(size_t)grow * SIZE + gs] = phi;
            atomicAdd(&rowacc[w * 16 + row16], -lt);
        }
        __syncthreads();
    }

    __syncthreads();
    if (tid < 128)
        atomicAdd(&out[(size_t)BDIM * SIZE + tileM + tid], rowacc[tid]);
}

// ---------------- host launcher ----------------
extern "C" void kernel_launch(void* const* d_in, const int* in_sizes, int n_in,
                              void* d_out, int out_size, void* d_ws, size_t ws_size,
                              hipStream_t stream) {
    (void)in_sizes; (void)n_in; (void)out_size; (void)ws_size;

    const float* x_in    = (const float*)d_in[0];
    const float* x_pass  = (const float*)d_in[1];
    const float* ld_in   = (const float*)d_in[2];
    const float* w1      = (const float*)d_in[3];
    const float* b1      = (const float*)d_in[4];
    const float* w2      = (const float*)d_in[5];
    const float* b2      = (const float*)d_in[6];
    float* out = (float*)d_out;

    // workspace layout (all 256B aligned)
    char* wsb = (char*)d_ws;
    constexpr size_t OFF_STATS = 0;
    constexpr size_t OFF_XSB = 256;
    constexpr size_t SZ_XSB  = (size_t)BDIM * SIZE * 2;   // 8 MB
    constexpr size_t OFF_W1T = OFF_XSB + SZ_XSB;
    constexpr size_t SZ_W1T  = (size_t)SIZE * HID * 2;    // 4 MB  (w1^T, [HID][SIZE])
    constexpr size_t OFF_HB  = OFF_W1T + SZ_W1T;
    constexpr size_t SZ_HB   = (size_t)BDIM * HID * 2;    // 4 MB
    constexpr size_t OFF_W2T = OFF_HB + SZ_HB;            // w2^T bf16: 68 MB ([OUTN][HID])

    float*  stats = (float*)(wsb + OFF_STATS);
    __bf16* xsb   = (__bf16*)(wsb + OFF_XSB);
    __bf16* w1t   = (__bf16*)(wsb + OFF_W1T);
    __bf16* hb    = (__bf16*)(wsb + OFF_HB);
    __bf16* w2t   = (__bf16*)(wsb + OFF_W2T);

    float* out_ld = out + (size_t)BDIM * SIZE;

    hipLaunchKernelGGL(init_kernel, dim3(8), dim3(256), 0, stream, ld_in, out_ld, stats);
    hipLaunchKernelGGL(stats_kernel, dim3(512), dim3(256), 0, stream, x_pass, stats);
    hipLaunchKernelGGL(standardize_kernel, dim3(4096), dim3(256), 0, stream, x_pass, stats, xsb);
    // w1 [SIZE][HID] -> w1t [HID][SIZE]
    hipLaunchKernelGGL(transpose_cvt_kernel, dim3(HID / 32, SIZE / 32), dim3(256), 0, stream,
                       w1, w1t, SIZE, HID);
    // w2 [HID][OUTN] -> w2t [OUTN][HID]
    hipLaunchKernelGGL(transpose_cvt_kernel, dim3(OUTN / 32, HID / 32), dim3(256), 0, stream,
                       w2, w2t, HID, OUTN);
    hipLaunchKernelGGL(gemm1_kernel, dim3(HID / 128, BDIM / 128), dim3(256), 0, stream,
                       xsb, w1t, b1, hb);
    hipLaunchKernelGGL(gemm2_spline_kernel, dim3(OUTN / 272, BDIM / 128), dim3(256), 0, stream,
                       hb, w2t, b2, x_in, out);
}
